// MaskedAttention_6811818132088
// MI455X (gfx1250) — matrix operand order repaired and checked by reading, not compiled, not run on hardware
//
#include <hip/hip_runtime.h>

// ---------------------------------------------------------------------------
// GPT-2 masked attention block for MI455X (gfx1250, wave32, WMMA + TDM).
// B=4, S=2048, NX=1024, H=16, hd=64.  All matmuls in bf16 WMMA 16x16x32,
// f32 accumulate.  GEMM tiles staged with tensor_load_to_lds (TDM, TENSORcnt,
// double-buffered LDS).  Flash-attention streaming softmax (no SxS in HBM).
// ---------------------------------------------------------------------------

typedef __bf16 bf16;
typedef __attribute__((ext_vector_type(8)))  bf16     v8bf;
typedef __attribute__((ext_vector_type(16))) bf16     v16bf;
typedef __attribute__((ext_vector_type(8)))  float    v8f;
typedef __attribute__((ext_vector_type(4)))  unsigned v4u;
typedef __attribute__((ext_vector_type(8)))  int      v8i;
typedef __attribute__((ext_vector_type(4)))  int      v4i;

union BFrag { v16bf v; v8bf h[2]; };

static constexpr int Bn = 4;
static constexpr int Sq = 2048;
static constexpr int NX = 1024;
static constexpr int Hn = 16;
static constexpr int HD = 64;
static constexpr int Mrows = Bn * Sq;       // 8192

__device__ __forceinline__ v8f wmma_bf16(const v16bf& a, const v16bf& b, const v8f& c) {
    return __builtin_amdgcn_wmma_f32_16x16x32_bf16(false, a, false, b, (short)0, c, false, false);
}

// ---------------------------------------------------------------------------
// TDM: DMA one [128 rows x 32 cols] bf16 tile (row pitch = stride elems)
// from global memory into LDS at byte offset lds_off.  2-D descriptor:
//   group0 = {count=1, lds_addr, gaddr[31:0], gaddr[56:32] | type(2)<<30}
//   group1 = {data_size=2B, tensor_dim0=32, tensor_dim1=128,
//             tile_dim0=32, tile_dim1=128, tensor_dim0_stride=stride}
// Issued by one wave; tracked with TENSORcnt.
// ---------------------------------------------------------------------------
__device__ __forceinline__ void tdm_load_tile_128x32(const bf16* gsrc, unsigned lds_off,
                                                     int stride_elems) {
    const unsigned long long ga = (unsigned long long)(size_t)gsrc;
    v4u g0 = { 1u,                                        // count=1 (valid D#)
               lds_off,                                   // LDS dest (bytes)
               (unsigned)(ga & 0xFFFFFFFFull),            // global_addr lo
               (unsigned)((ga >> 32) & 0x1FFFFFFull) | 0x80000000u }; // hi | type=2
    v8i g1 = { (int)0x10000,                              // data_size = 2 bytes
               (int)(32u << 16),                          // tensor_dim0 = 32
               (int)(128u << 16),                         // tensor_dim1 = 128
               (int)(32u << 16),                          // tile_dim0   = 32
               (int)128,                                  // tile_dim1   = 128
               stride_elems,                              // tensor_dim0_stride lo
               0, 0 };
    v4i gz = { 0, 0, 0, 0 };
#if __clang_major__ >= 23
    v8i gz8 = { 0, 0, 0, 0, 0, 0, 0, 0 };
    __builtin_amdgcn_tensor_load_to_lds(g0, g1, gz, gz, gz8, 0);
#else
    __builtin_amdgcn_tensor_load_to_lds(g0, g1, gz, gz, 0);
#endif
}

// ---------------------------------------------------------------------------
// fp32 -> bf16 straight convert
// ---------------------------------------------------------------------------
__global__ __launch_bounds__(256)
void cvt_kernel(const float* __restrict__ in, bf16* __restrict__ out, int n) {
    int i = blockIdx.x * 256 + threadIdx.x;
    if (i < n) out[i] = (bf16)in[i];
}

// fp32 [K][N] -> bf16 [N][K] (transpose so GEMM B-tiles are [n][k] in LDS)
__global__ __launch_bounds__(256)
void cvt_t_kernel(const float* __restrict__ in, bf16* __restrict__ out, int K, int N) {
    long long i = (long long)blockIdx.x * 256 + threadIdx.x;
    if (i < (long long)K * N) {
        int k = (int)(i / N);
        int n = (int)(i % N);
        out[(long long)n * K + k] = (bf16)in[i];
    }
}

// ---------------------------------------------------------------------------
// Tiled GEMM mainloop (shared by QKV + proj kernels via macro):
// C[128x128] per block, 8 waves, wave tile 64x32, K-step 32.
// A: [M][K] bf16 row-major.  Bt: [N][K] bf16 row-major (pre-transposed).
// Staging: wave 0 issues TDM loads for tile j+1 into the alternate LDS
// buffer while all waves compute on tile j; s_wait_tensorcnt + barrier
// hands the DMA'd tile to the consumers.
// ---------------------------------------------------------------------------
#define GEMM_MAINLOOP(A_, Bt_, Ktot_)                                          \
    __shared__ __align__(16) bf16 lsA[2][128 * 32];                            \
    __shared__ __align__(16) bf16 lsB[2][128 * 32];                            \
    const int t    = threadIdx.x;                                              \
    const int lane = t & 31;                                                   \
    const int wave = t >> 5;                                                   \
    const int wm   = wave >> 2;        /* 0..1 */                              \
    const int wn   = wave & 3;         /* 0..3 */                              \
    const int m0   = blockIdx.x * 128;                                         \
    const int n0   = blockIdx.y * 128;                                         \
    const int fm   = lane & 15;                                                \
    const int fhl  = lane >> 4;                                                \
    v8f acc[4][2];                                                             \
    _Pragma("unroll") for (int i_ = 0; i_ < 4; ++i_)                           \
        _Pragma("unroll") for (int j_ = 0; j_ < 2; ++j_) acc[i_][j_] = {};     \
    const bf16* gA_ = (A_)  + (long long)m0 * (Ktot_);                         \
    const bf16* gB_ = (Bt_) + (long long)n0 * (Ktot_);                         \
    const int nk_ = (Ktot_) / 32;                                              \
    if (wave == 0) {                                                           \
        tdm_load_tile_128x32(gA_, (unsigned)(size_t)&lsA[0][0], (Ktot_));      \
        tdm_load_tile_128x32(gB_, (unsigned)(size_t)&lsB[0][0], (Ktot_));      \
        __builtin_amdgcn_s_wait_tensorcnt((short)0);                           \
    }                                                                          \
    __syncthreads();                                                           \
    for (int j_ = 0; j_ < nk_; ++j_) {                                         \
        const int buf_ = j_ & 1;                                               \
        if (wave == 0 && (j_ + 1) < nk_) {                                     \
            tdm_load_tile_128x32(gA_ + (j_ + 1) * 32,                          \
                                 (unsigned)(size_t)&lsA[buf_ ^ 1][0], (Ktot_));\
            tdm_load_tile_128x32(gB_ + (j_ + 1) * 32,                          \
                                 (unsigned)(size_t)&lsB[buf_ ^ 1][0], (Ktot_));\
        }                                                                      \
        const bf16* la_ = lsA[buf_];                                           \
        const bf16* lb_ = lsB[buf_];                                           \
        BFrag bfr[2];                                                          \
        _Pragma("unroll") for (int nt = 0; nt < 2; ++nt) {                     \
            const int nr = wn * 32 + nt * 16 + fm;                             \
            bfr[nt].h[0] = *(const v8bf*)&lb_[nr * 32 + fhl * 8];              \
            bfr[nt].h[1] = *(const v8bf*)&lb_[nr * 32 + 16 + fhl * 8];         \
        }                                                                      \
        _Pragma("unroll") for (int mt = 0; mt < 4; ++mt) {                     \
            BFrag af;                                                          \
            const int mr = wm * 64 + mt * 16 + fm;                             \
            af.h[0] = *(const v8bf*)&la_[mr * 32 + fhl * 8];                   \
            af.h[1] = *(const v8bf*)&la_[mr * 32 + 16 + fhl * 8];              \
            _Pragma("unroll") for (int nt = 0; nt < 2; ++nt)                   \
                acc[mt][nt] = wmma_bf16(af.v, bfr[nt].v, acc[mt][nt]);         \
        }                                                                      \
        if (wave == 0) __builtin_amdgcn_s_wait_tensorcnt((short)0);            \
        __syncthreads();                                                       \
    }

// ---------------------------------------------------------------------------
// QKV GEMM: [8192 x 1024] @ [1024 x 3072] + bias.
// Epilogue scatters bf16 into head-major Q [BH,S,64], K [BH,S,64],
// and V transposed [BH,64,S] (so P@V B-fragments are contiguous).
// ---------------------------------------------------------------------------
__global__ __launch_bounds__(256)
void gemm_qkv_kernel(const bf16* __restrict__ A, const bf16* __restrict__ Bt,
                     const float* __restrict__ bias,
                     bf16* __restrict__ Qb, bf16* __restrict__ Kb, bf16* __restrict__ Vb) {
    GEMM_MAINLOOP(A, Bt, NX)
#pragma unroll
    for (int mt = 0; mt < 4; ++mt)
#pragma unroll
        for (int nt = 0; nt < 2; ++nt)
#pragma unroll
            for (int r = 0; r < 8; ++r) {
                const int gm = m0 + wm * 64 + mt * 16 + r + fhl * 8;
                const int gn = n0 + wn * 32 + nt * 16 + fm;
                const float v = acc[mt][nt][r] + bias[gn];
                const int b = gm >> 11;          // /2048
                const int s = gm & 2047;
                const int which = gn >> 10;      // 0=Q 1=K 2=V
                const int nn = gn & 1023;
                const int h = nn >> 6, d = nn & 63;
                const long long bh = (long long)(b * Hn + h);
                if (which == 0)      Qb[(bh * Sq + s) * HD + d] = (bf16)v;
                else if (which == 1) Kb[(bh * Sq + s) * HD + d] = (bf16)v;
                else                 Vb[(bh * HD + d) * Sq + s] = (bf16)v;
            }
}

// ---------------------------------------------------------------------------
// Output projection GEMM: [8192 x 1024] @ [1024 x 1024] + bias -> fp32 out.
// ---------------------------------------------------------------------------
__global__ __launch_bounds__(256)
void gemm_proj_kernel(const bf16* __restrict__ A, const bf16* __restrict__ Bt,
                      const float* __restrict__ bias, float* __restrict__ out) {
    GEMM_MAINLOOP(A, Bt, NX)
#pragma unroll
    for (int mt = 0; mt < 4; ++mt)
#pragma unroll
        for (int nt = 0; nt < 2; ++nt)
#pragma unroll
            for (int r = 0; r < 8; ++r) {
                const int gm = m0 + wm * 64 + mt * 16 + r + fhl * 8;
                const int gn = n0 + wn * 32 + nt * 16 + fm;
                out[(long long)gm * NX + gn] = acc[mt][nt][r] + bias[gn];
            }
}

// ---------------------------------------------------------------------------
// Flash attention, causal, per (b,h).  Block = 4 waves = 64 q-rows; each
// wave owns a 16-row q tile.  kv streamed in blocks of 32 (K,V staged in
// LDS, shared by all 4 waves).  Online softmax; P re-layout C-frag->A-frag
// via per-wave LDS round trip.
// ---------------------------------------------------------------------------
__global__ __launch_bounds__(128)
void attn_kernel(const bf16* __restrict__ Qb, const bf16* __restrict__ Kb,
                 const bf16* __restrict__ Vb, bf16* __restrict__ Ab) {
    __shared__ __align__(16) bf16 lsK[32 * 64];       // [kv][d]
    __shared__ __align__(16) bf16 lsV[64 * 32];       // [d][kv]  (V transposed)
    __shared__ __align__(16) bf16 lsP[4][16 * 32];    // per-wave P tile

    const int t    = threadIdx.x;
    const int lane = t & 31;
    const int wave = t >> 5;
    const int fm   = lane & 15;
    const int fhl  = lane >> 4;
    const int bh   = blockIdx.y;                      // 0..63  (b*16 + h)
    const int q0   = blockIdx.x * 64;
    const int qw   = q0 + wave * 16;

    const bf16* Qh = Qb + (long long)bh * Sq * HD;
    const bf16* Kh = Kb + (long long)bh * Sq * HD;
    const bf16* Vh = Vb + (long long)bh * HD * Sq;

    // Q A-fragments for 2 K-steps (d = 0..31, 32..63), resident whole kernel
    BFrag qf[2];
#pragma unroll
    for (int ks = 0; ks < 2; ++ks) {
        const bf16* p = Qh + (long long)(qw + fm) * HD + ks * 32 + fhl * 8;
        qf[ks].h[0] = *(const v8bf*)(p);
        qf[ks].h[1] = *(const v8bf*)(p + 16);
    }

    v8f o[4];
#pragma unroll
    for (int dt = 0; dt < 4; ++dt) o[dt] = {};
    float mrow[8], lrow[8];
#pragma unroll
    for (int r = 0; r < 8; ++r) { mrow[r] = -1e30f; lrow[r] = 0.0f; }

    const int kv_end = q0 + 64;                       // causal horizon for block
    for (int kv0 = 0; kv0 < kv_end; kv0 += 32) {
        __syncthreads();
        // stage K block [32][64] : contiguous 4KB from global
        {
            const v8bf* src = (const v8bf*)(Kh + (long long)kv0 * HD);
            v8bf* dst = (v8bf*)lsK;
            dst[t]       = src[t];
            dst[t + 128] = src[t + 128];
        }
        // stage V block [64 d][32 kv] from Vb[d][S]
        {
            const int d = t >> 1, hf = t & 1;
            const bf16* vs = Vh + (long long)d * Sq + kv0 + hf * 16;
            *(v8bf*)&lsV[d * 32 + hf * 16]     = *(const v8bf*)(vs);
            *(v8bf*)&lsV[d * 32 + hf * 16 + 8] = *(const v8bf*)(vs + 8);
        }
        __syncthreads();

        // scores: Q[16x64] @ K^T -> two 16x16 tiles, accumulate 2 K-steps each
        v8f sc[2];
#pragma unroll
        for (int nt = 0; nt < 2; ++nt) {
            sc[nt] = {};
#pragma unroll
            for (int ks = 0; ks < 2; ++ks) {
                BFrag kf;
                const int nr = nt * 16 + fm;          // kv row within block
                kf.h[0] = *(const v8bf*)&lsK[nr * 64 + ks * 32 + fhl * 8];
                kf.h[1] = *(const v8bf*)&lsK[nr * 64 + ks * 32 + 16 + fhl * 8];
                sc[nt] = wmma_bf16(qf[ks].v, kf.v, sc[nt]);
            }
        }

        // causal mask + scale + online softmax (row stats across 16 lanes)
#pragma unroll
        for (int r = 0; r < 8; ++r) {
            const int qrow = qw + r + fhl * 8;
            float s0 = sc[0][r] * 0.125f;             // 1/sqrt(64)
            float s1 = sc[1][r] * 0.125f;
            s0 = (kv0 + fm      <= qrow) ? s0 : -10000.0f;
            s1 = (kv0 + 16 + fm <= qrow) ? s1 : -10000.0f;
            float mx = fmaxf(s0, s1);
#pragma unroll
            for (int msk = 8; msk >= 1; msk >>= 1)
                mx = fmaxf(mx, __shfl_xor(mx, msk, 32));
            const float mnew  = fmaxf(mrow[r], mx);
            const float alpha = __expf(mrow[r] - mnew);
            const float p0 = __expf(s0 - mnew);
            const float p1 = __expf(s1 - mnew);
            float ps = p0 + p1;
#pragma unroll
            for (int msk = 8; msk >= 1; msk >>= 1)
                ps += __shfl_xor(ps, msk, 32);
            lrow[r] = lrow[r] * alpha + ps;
            mrow[r] = mnew;
#pragma unroll
            for (int dt = 0; dt < 4; ++dt) o[dt][r] = o[dt][r] * alpha;
            sc[0][r] = p0; sc[1][r] = p1;
        }

        // C-fragment -> A-fragment re-layout of P via per-wave LDS tile
        bf16* P = lsP[wave];
#pragma unroll
        for (int r = 0; r < 8; ++r) {
            const int rr = r + fhl * 8;
            P[rr * 32 + fm]      = (bf16)sc[0][r];
            P[rr * 32 + 16 + fm] = (bf16)sc[1][r];
        }
        asm volatile("s_wait_dscnt 0" ::: "memory");   // wave-local LDS RAW fence
        BFrag pf;
        pf.h[0] = *(const v8bf*)&P[fm * 32 + fhl * 8];
        pf.h[1] = *(const v8bf*)&P[fm * 32 + 16 + fhl * 8];

        // O += P[16x32] @ V[32x64]  (one K=32 wmma per 16-wide d tile)
#pragma unroll
        for (int dt = 0; dt < 4; ++dt) {
            BFrag vf;
            const int dr = dt * 16 + fm;
            vf.h[0] = *(const v8bf*)&lsV[dr * 32 + fhl * 8];
            vf.h[1] = *(const v8bf*)&lsV[dr * 32 + 16 + fhl * 8];
            o[dt] = wmma_bf16(pf.v, vf.v, o[dt]);
        }
    }

    // normalize and store merged-head bf16 activations [B,S,NX]
    const int b = bh >> 4, h = bh & 15;
#pragma unroll
    for (int dt = 0; dt < 4; ++dt)
#pragma unroll
        for (int r = 0; r < 8; ++r) {
            const int qrow = qw + r + fhl * 8;
            const float v = o[dt][r] / lrow[r];
            Ab[((long long)(b * Sq + qrow)) * NX + h * HD + dt * 16 + fm] = (bf16)v;
        }
}

// ---------------------------------------------------------------------------
// Host launcher
// ---------------------------------------------------------------------------
extern "C" void kernel_launch(void* const* d_in, const int* in_sizes, int n_in,
                              void* d_out, int out_size, void* d_ws, size_t ws_size,
                              hipStream_t stream) {
    const float* x        = (const float*)d_in[0];
    const float* c_attn_w = (const float*)d_in[1];
    const float* c_attn_b = (const float*)d_in[2];
    const float* c_proj_w = (const float*)d_in[3];
    const float* c_proj_b = (const float*)d_in[4];
    float* out = (float*)d_out;

    // workspace layout (bf16, all 16B-aligned offsets)
    char* ws = (char*)d_ws;
    bf16* xb     = (bf16*)ws;                           ws += (long long)Mrows * NX * 2;      // 16.8MB
    bf16* wqkvT  = (bf16*)ws;                           ws += (long long)3 * NX * NX * 2;     //  6.3MB
    bf16* wprojT = (bf16*)ws;                           ws += (long long)NX * NX * 2;         //  2.1MB
    bf16* Qb     = (bf16*)ws;                           ws += (long long)Bn * Hn * Sq * HD * 2;
    bf16* Kb     = (bf16*)ws;                           ws += (long long)Bn * Hn * Sq * HD * 2;
    bf16* Vb     = (bf16*)ws;                           ws += (long long)Bn * Hn * Sq * HD * 2;
    bf16* Ab     = (bf16*)ws;                           ws += (long long)Mrows * NX * 2;
    // total ~92MB

    // 1) precision conversion (+ weight transpose to [N][K])
    {
        int n = Mrows * NX;                               // 8,388,608
        cvt_kernel<<<(n + 255) / 256, 256, 0, stream>>>(x, xb, n);
        int na = NX * 3 * NX;                             // 3,145,728
        cvt_t_kernel<<<(na + 255) / 256, 256, 0, stream>>>(c_attn_w, wqkvT, NX, 3 * NX);
        int np = NX * NX;                                 // 1,048,576
        cvt_t_kernel<<<(np + 255) / 256, 256, 0, stream>>>(c_proj_w, wprojT, NX, NX);
    }

    // 2) fused QKV projection -> head-major Q/K/V (V transposed)
    gemm_qkv_kernel<<<dim3(Mrows / 128, (3 * NX) / 128), 256, 0, stream>>>(
        xb, wqkvT, c_attn_b, Qb, Kb, Vb);

    // 3) causal flash attention per (b,h)
    attn_kernel<<<dim3(Sq / 64, Bn * Hn), 128, 0, stream>>>(Qb, Kb, Vb, Ab);

    // 4) output projection -> fp32
    gemm_proj_kernel<<<dim3(Mrows / 128, NX / 128), 256, 0, stream>>>(
        Ab, wprojT, c_proj_b, out);
}
